// Mamba3Block_13116830122043
// MI455X (gfx1250) — matrix operand order repaired and checked by reading, not compiled, not run on hardware
//
#include <hip/hip_runtime.h>
#include <hip/hip_bf16.h>
#include <math.h>

// ---------------- problem constants ----------------
#define SEQ      1024
#define DMODEL   768
#define DINNER   1536
#define DSTATE   192
#define NHEADS   24
#define HEADDIM  64
#define NPAIR    48
#define NROT     96
#define DINPROJ  3480      // 2*DINNER + 2*DSTATE + NHEADS
#define NPAD1    3584      // DINPROJ padded up to 28*128 for guard-free GEMM1 B reads
#define EPSF     1e-6f
#define AFLOOR   1e-4f

// offsets inside a zxbcdt row
#define OFF_Z    0
#define OFF_U    1536
#define OFF_B    3072
#define OFF_C    3264
#define OFF_DT   3456

typedef __bf16 v16bf __attribute__((ext_vector_type(16)));
typedef float  v8f   __attribute__((ext_vector_type(8)));
typedef unsigned int u32x4 __attribute__((ext_vector_type(4)));
typedef unsigned int u32x8 __attribute__((ext_vector_type(8)));

static __device__ __forceinline__ __bf16 f2bf(float f) {
    unsigned u = __float_as_uint(f);
    u += 0x7FFFu + ((u >> 16) & 1u);            // round-to-nearest-even
    unsigned short hs = (unsigned short)(u >> 16);
    __bf16 r;
    __builtin_memcpy(&r, &hs, sizeof(r));
    return r;
}

// ---- Tensor Data Mover: DMA a 32(col, bf16) x 128(row) tile into LDS ----
// LDS destination rows are padded to 40 bf16 (80 B) via TDM pad feature:
// pad_interval=3 (insert after 16 DWORDs = 64 B), pad_amount=3 (4 DWORDs = 16 B).
static __device__ __forceinline__ void tdm_load_a_tile(const __bf16* gsrc,
                                                       unsigned lds_byte_addr,
                                                       int lda_elems) {
    unsigned long long ga = (unsigned long long)(size_t)gsrc;
    u32x4 g0;
    g0[0] = 1u;                                            // count=1, no gather
    g0[1] = lds_byte_addr;                                 // D#.lds_addr
    g0[2] = (unsigned)ga;                                  // global_addr[31:0]
    g0[3] = (unsigned)((ga >> 32) & 0x01FFFFFFull)         // global_addr[56:32]
          | (2u << 30);                                    // type = 2 ("image")
    u32x8 g1;
    g1[0] = (1u << 16)                                     // data_size = 2 bytes
          | (1u << 20)                                     // pad_enable
          | (3u << 22)                                     // pad_interval: 16 DWORDs
          | (3u << 25);                                    // pad_amount: 4 DWORDs
    g1[1] = (32u & 0xFFFFu) << 16;                         // tensor_dim0 = 32 (lo16)
    g1[2] = (128u & 0xFFFFu) << 16;                        // tdim0 hi | tensor_dim1 = 128 (lo16)
    g1[3] = (32u << 16);                                   // tdim1 hi | tile_dim0 = 32
    g1[4] = 128u;                                          // tile_dim1 = 128, tile_dim2 = 0
    g1[5] = (unsigned)lda_elems;                           // tensor_dim0_stride lo32
    g1[6] = 0u;                                            // stride hi | dim1_stride lo
    g1[7] = 0u;
    asm volatile("tensor_load_to_lds %0, %1" :: "s"(g0), "s"(g1) : "memory");
}

// ---------------- K0: weight packing to bf16 ----------------
__global__ void pack_win_kernel(const float* __restrict__ win, __bf16* __restrict__ dst) {
    int idx = blockIdx.x * 256 + threadIdx.x;           // over 768*3584
    if (idx >= DMODEL * NPAD1) return;
    int col = idx % NPAD1, row = idx / NPAD1;
    float v = (col < DINPROJ) ? win[row * DINPROJ + col] : 0.0f;
    dst[idx] = f2bf(v);
}

__global__ void pack_wout_kernel(const float* __restrict__ wout, __bf16* __restrict__ dst) {
    int idx = blockIdx.x * 256 + threadIdx.x;           // over 1536*768
    if (idx >= DINNER * DMODEL) return;
    dst[idx] = f2bf(wout[idx]);
}

// ---------------- K1: pre RMS-norm -> bf16 h ----------------
__global__ void prenorm_kernel(const float* __restrict__ x, const float* __restrict__ pre_scale,
                               __bf16* __restrict__ h) {
    __shared__ float red[256];
    int t = blockIdx.x, tid = threadIdx.x;
    const float* row = x + (size_t)t * DMODEL;
    float v[3], ss = 0.f;
    for (int i = 0; i < 3; ++i) { v[i] = row[tid + i * 256]; ss += v[i] * v[i]; }
    red[tid] = ss; __syncthreads();
    for (int s = 128; s > 0; s >>= 1) { if (tid < s) red[tid] += red[tid + s]; __syncthreads(); }
    float scale = rsqrtf(red[0] / (float)DMODEL + EPSF);
    for (int i = 0; i < 3; ++i) {
        int c = tid + i * 256;
        h[(size_t)t * DMODEL + c] = f2bf(v[i] * scale * (1.0f + pre_scale[c]));
    }
}

// ---------------- WMMA GEMM: C[MxN] = A[MxK](bf16,row) * B[KxN](bf16,row) ----------------
// block tile 128x128, K staged 32 wide, double-buffered LDS.
// A tiles staged by the Tensor Data Mover (wave 0 issues, TENSORcnt handoff);
// B tiles staged by all threads (coalesced b128 global loads, transposed spill to LDS).
// 8 waves as 4(M) x 2(N), each wave owns 2x4 16x16 tiles -> 8 WMMA per K-step.
__global__ void __launch_bounds__(256)
wmma_gemm_bf16(const __bf16* __restrict__ A, const __bf16* __restrict__ B,
               float* __restrict__ C, int M, int N, int K, int lda, int ldb, int ldc) {
    __shared__ __align__(16) __bf16 Alds[2][128][40];   // [buf][m][k], TDM-padded rows
    __shared__ __align__(16) __bf16 Blds[2][128][40];   // [buf][n][k], transposed

    int tid = threadIdx.x;
    int m0 = blockIdx.y * 128;
    int n0 = blockIdx.x * 128;
    int lane = tid & 31, half = lane >> 4, ml = lane & 15;
    int w = tid >> 5, wm = w & 3, wn = w >> 2;          // wave: 32 M-rows, 64 N-cols

    v8f acc[2][4] = {};

    // ---- preload tile 0 ----
    if (w == 0)
        tdm_load_a_tile(A + (size_t)m0 * lda,
                        (unsigned)(size_t)&Alds[0][0][0], lda);
    {
        int kk = tid >> 4;
        int ng = (tid & 15) * 8;
        #pragma unroll
        for (int it = 0; it < 2; ++it) {
            uint4 v = *(const uint4*)(B + (size_t)(it * 16 + kk) * ldb + n0 + ng);
            __bf16 tmp[8]; __builtin_memcpy(tmp, &v, 16);
            #pragma unroll
            for (int i = 0; i < 8; ++i) Blds[0][ng + i][it * 16 + kk] = tmp[i];
        }
    }
    if (w == 0) __builtin_amdgcn_s_wait_tensorcnt(0);
    __syncthreads();

    int nsteps = K / 32;                                 // K is a multiple of 32 here
    for (int s = 0; s < nsteps; ++s) {
        int cur = s & 1, nxt = cur ^ 1;
        int k0 = s * 32;
        bool more = (s + 1 < nsteps);
        int kk = tid >> 4;
        int ng = (tid & 15) * 8;

        uint4 bv[2];
        if (more) {
            // next A tile via TDM into the alternate buffer
            if (w == 0)
                tdm_load_a_tile(A + (size_t)m0 * lda + (k0 + 32),
                                (unsigned)(size_t)&Alds[nxt][0][0], lda);
            // next B tile: global loads in flight across the WMMA phase
            #pragma unroll
            for (int it = 0; it < 2; ++it)
                bv[it] = *(const uint4*)(B + (size_t)(k0 + 32 + it * 16 + kk) * ldb + n0 + ng);
            if (s + 2 < nsteps)      // two tiles ahead (global_prefetch_b8)
                __builtin_prefetch(B + (size_t)(k0 + 64 + kk) * ldb + n0 + ng, 0, 3);
        }

        // fragments per documented wave32 VGPR layouts
        v16bf afrag[2], bfrag[4];
        #pragma unroll
        for (int i = 0; i < 2; ++i) {
            int row = wm * 32 + i * 16 + ml;
            #pragma unroll
            for (int j = 0; j < 8; ++j) {
                afrag[i][j]     = Alds[cur][row][half * 8 + j];
                afrag[i][j + 8] = Alds[cur][row][16 + half * 8 + j];
            }
        }
        #pragma unroll
        for (int jt = 0; jt < 4; ++jt) {
            int col = wn * 64 + jt * 16 + ml;
            #pragma unroll
            for (int e = 0; e < 16; ++e) bfrag[jt][e] = Blds[cur][col][half * 8 + e + (e & 8)];
        }
        #pragma unroll
        for (int i = 0; i < 2; ++i)
            #pragma unroll
            for (int jt = 0; jt < 4; ++jt)
                acc[i][jt] = __builtin_amdgcn_wmma_f32_16x16x32_bf16(
                    false, afrag[i], false, bfrag[jt], (short)0, acc[i][jt], false, false);

        if (more) {
            #pragma unroll
            for (int it = 0; it < 2; ++it) {
                __bf16 tmp[8]; __builtin_memcpy(tmp, &bv[it], 16);
                #pragma unroll
                for (int i = 0; i < 8; ++i) Blds[nxt][ng + i][it * 16 + kk] = tmp[i];
            }
            if (w == 0) __builtin_amdgcn_s_wait_tensorcnt(0);
        }
        __syncthreads();
    }

    // store D (f32 16x16 layout: n = ml, m = half*8 + r)
    #pragma unroll
    for (int i = 0; i < 2; ++i)
        #pragma unroll
        for (int jt = 0; jt < 4; ++jt) {
            int gn = n0 + wn * 64 + jt * 16 + ml;
            if (gn < N) {
                int gmb = m0 + wm * 32 + i * 16 + half * 8;
                #pragma unroll
                for (int r = 0; r < 8; ++r) {
                    int gm = gmb + r;
                    if (gm < M) C[(size_t)gm * ldc + gn] = acc[i][jt][r];
                }
            }
        }
}

// ---------------- K3: per-head dt / a / cumsum(dt) ----------------
__global__ void dtprep_kernel(const float* __restrict__ zx, const float* __restrict__ dt_bias,
                              const float* __restrict__ A_log,
                              float* __restrict__ dtg, float* __restrict__ ag,
                              float* __restrict__ cumg) {
    __shared__ float sdt[SEQ];
    int h = blockIdx.x, tid = threadIdx.x;
    float Ah = -(expf(A_log[h]) + AFLOOR);
    float bias = dt_bias[h];
    for (int t = tid; t < SEQ; t += 256) {
        float xv = zx[(size_t)t * DINPROJ + OFF_DT + h] + bias;
        float dt = (xv > 20.0f) ? xv : log1pf(expf(xv));   // softplus
        sdt[t] = dt;
        dtg[h * SEQ + t] = dt;
        ag[h * SEQ + t]  = expf(dt * Ah);
    }
    __syncthreads();
    // Hillis-Steele inclusive scan over 1024 elements, 4 per thread
    for (int off = 1; off < SEQ; off <<= 1) {
        float vals[4];
        #pragma unroll
        for (int i = 0; i < 4; ++i) {
            int idx = tid + i * 256;
            vals[i] = (idx >= off) ? sdt[idx - off] : 0.0f;
        }
        __syncthreads();
        #pragma unroll
        for (int i = 0; i < 4; ++i) sdt[tid + i * 256] += vals[i];
        __syncthreads();
    }
    for (int t = tid; t < SEQ; t += 256) cumg[h * SEQ + t] = sdt[t];
}

// ---------------- K4: the serial SSM scan, state register-resident ----------------
// one block per head; 256 threads; thread (p = tid>>2) owns HEADDIM row p,
// q = tid&3 owns 48 of the 192 state columns -> 48 f32 regs of state per lane.
__global__ void __launch_bounds__(256)
scan_kernel(const float* __restrict__ zx, const float* __restrict__ dtg,
            const float* __restrict__ ag, const float* __restrict__ cumg,
            const float* __restrict__ omega, const float* __restrict__ Dp,
            float* __restrict__ yws) {
    __shared__ float Bbuf[2][DSTATE];
    __shared__ float ubuf[2][HEADDIM];
    __shared__ float Cbuf[DSTATE];
    __shared__ float om[NPAIR];

    int h = blockIdx.x, tid = threadIdx.x;
    int q = tid & 3, p = tid >> 2;

    if (tid < NPAIR)  om[tid] = omega[h * NPAIR + tid];
    if (tid < DSTATE) Bbuf[1][tid] = 0.0f;   // t=0 "prev" is zero
    if (tid < HEADDIM) ubuf[1][tid] = 0.0f;
    float Dh = Dp[h];
    float s[48];
    #pragma unroll
    for (int k = 0; k < 48; ++k) s[k] = 0.0f;
    __syncthreads();

    for (int t = 0; t < SEQ; ++t) {
        int cur = t & 1;
        const float* row = zx + (size_t)t * DINPROJ;

        // stage + rotate B (cos,-sin) and C (cos,sin); copy n>=NROT; stage u
        if (tid < NPAIR) {
            float phi = cumg[h * SEQ + t] * om[tid];
            float sn, cs;
            __sincosf(phi, &sn, &cs);
            float b0 = row[OFF_B + 2 * tid], b1 = row[OFF_B + 2 * tid + 1];
            Bbuf[cur][2 * tid]     =  b0 * cs + b1 * sn;
            Bbuf[cur][2 * tid + 1] = -b0 * sn + b1 * cs;
            float c0 = row[OFF_C + 2 * tid], c1 = row[OFF_C + 2 * tid + 1];
            Cbuf[2 * tid]     = c0 * cs - c1 * sn;
            Cbuf[2 * tid + 1] = c0 * sn + c1 * cs;
        } else if (tid < 96) {
            int n = NROT + (tid - NPAIR) * 2;
            Bbuf[cur][n]     = row[OFF_B + n];
            Bbuf[cur][n + 1] = row[OFF_B + n + 1];
            Cbuf[n]          = row[OFF_C + n];
            Cbuf[n + 1]      = row[OFF_C + n + 1];
        } else if (tid < 160) {
            int pp = tid - 96;
            ubuf[cur][pp] = row[OFF_U + h * HEADDIM + pp];
        }
        __syncthreads();

        float at  = ag[h * SEQ + t];
        float dtt = dtg[h * SEQ + t];
        float c1 = 0.5f * dtt, c2 = at * c1;
        float f1 = c1 * ubuf[cur][p];
        float f2 = c2 * ubuf[cur ^ 1][p];
        const float* Bc = Bbuf[cur];
        const float* Bp = Bbuf[cur ^ 1];
        int nb = q * 48;
        float acc = 0.0f;
        #pragma unroll
        for (int k = 0; k < 48; ++k) {
            float sv = at * s[k] + f1 * Bc[nb + k] + f2 * Bp[nb + k];
            s[k] = sv;
            acc += sv * Cbuf[nb + k];
        }
        acc += __shfl_xor(acc, 1);
        acc += __shfl_xor(acc, 2);
        if (q == 0)
            yws[(size_t)t * DINNER + h * HEADDIM + p] = acc + Dh * ubuf[cur][p];
        __syncthreads();   // protect prev buffers / Cbuf before next staging
    }
}

// ---------------- K5: SiLU gate + RMS-norm * weight -> bf16 g ----------------
__global__ void gate_norm_kernel(const float* __restrict__ zx, const float* __restrict__ yws,
                                 const float* __restrict__ nw, __bf16* __restrict__ g) {
    __shared__ float red[256];
    int t = blockIdx.x, tid = threadIdx.x;
    const float* zr = zx + (size_t)t * DINPROJ;        // z at offset 0
    const float* yr = yws + (size_t)t * DINNER;
    float gv[6], ss = 0.f;
    #pragma unroll
    for (int i = 0; i < 6; ++i) {
        int c = tid + i * 256;
        float z = zr[c];
        float silu = z / (1.0f + expf(-z));
        float v = yr[c] * silu;
        gv[i] = v; ss += v * v;
    }
    red[tid] = ss; __syncthreads();
    for (int s2 = 128; s2 > 0; s2 >>= 1) { if (tid < s2) red[tid] += red[tid + s2]; __syncthreads(); }
    float scale = rsqrtf(red[0] / (float)DINNER + EPSF);
    #pragma unroll
    for (int i = 0; i < 6; ++i) {
        int c = tid + i * 256;
        g[(size_t)t * DINNER + c] = f2bf(gv[i] * scale * nw[c]);
    }
}

// ---------------- K6: post RMS-norm + residual ----------------
__global__ void postnorm_kernel(const float* __restrict__ y2, const float* __restrict__ x,
                                const float* __restrict__ post_scale, float* __restrict__ out) {
    __shared__ float red[256];
    int t = blockIdx.x, tid = threadIdx.x;
    const float* yr = y2 + (size_t)t * DMODEL;
    float v[3], ss = 0.f;
    #pragma unroll
    for (int i = 0; i < 3; ++i) { v[i] = yr[tid + i * 256]; ss += v[i] * v[i]; }
    red[tid] = ss; __syncthreads();
    for (int s = 128; s > 0; s >>= 1) { if (tid < s) red[tid] += red[tid + s]; __syncthreads(); }
    float scale = rsqrtf(red[0] / (float)DMODEL + EPSF);
    #pragma unroll
    for (int i = 0; i < 3; ++i) {
        int c = tid + i * 256;
        out[(size_t)t * DMODEL + c] = x[(size_t)t * DMODEL + c]
                                    + v[i] * scale * (1.0f + post_scale[c]);
    }
}

// ---------------- launcher ----------------
extern "C" void kernel_launch(void* const* d_in, const int* in_sizes, int n_in,
                              void* d_out, int out_size, void* d_ws, size_t ws_size,
                              hipStream_t stream) {
    const float* x          = (const float*)d_in[0];
    const float* pre_scale  = (const float*)d_in[1];
    const float* W_in       = (const float*)d_in[2];
    const float* dt_bias    = (const float*)d_in[3];
    const float* A_log      = (const float*)d_in[4];
    const float* omega      = (const float*)d_in[5];
    const float* Dp         = (const float*)d_in[6];
    const float* nw         = (const float*)d_in[7];
    const float* W_out      = (const float*)d_in[8];
    const float* post_scale = (const float*)d_in[9];
    float* out = (float*)d_out;

    char* ws = (char*)d_ws;
    size_t off = 0;
    auto alloc = [&](size_t bytes) -> void* {
        void* p = ws + off;
        off = (off + bytes + 255) & ~(size_t)255;
        return p;
    };
    __bf16* WinB  = (__bf16*)alloc((size_t)DMODEL * NPAD1 * 2);
    __bf16* WoutB = (__bf16*)alloc((size_t)DINNER * DMODEL * 2);
    __bf16* hB    = (__bf16*)alloc((size_t)SEQ * DMODEL * 2);
    float*  zx    = (float*)alloc((size_t)SEQ * DINPROJ * 4);
    float*  dtg   = (float*)alloc((size_t)NHEADS * SEQ * 4);
    float*  ag    = (float*)alloc((size_t)NHEADS * SEQ * 4);
    float*  cumg  = (float*)alloc((size_t)NHEADS * SEQ * 4);
    float*  yws   = (float*)alloc((size_t)SEQ * DINNER * 4);
    __bf16* gB    = (__bf16*)alloc((size_t)SEQ * DINNER * 2);
    float*  y2    = (float*)alloc((size_t)SEQ * DMODEL * 4);

    // 0) weight packing (fp32 -> bf16, W_in zero-padded to 3584 cols)
    pack_win_kernel<<<(DMODEL * NPAD1 + 255) / 256, 256, 0, stream>>>(W_in, WinB);
    pack_wout_kernel<<<(DINNER * DMODEL + 255) / 256, 256, 0, stream>>>(W_out, WoutB);
    // 1) pre-norm
    prenorm_kernel<<<SEQ, 256, 0, stream>>>(x, pre_scale, hB);
    // 2) GEMM1: zx[1024x3480] = hB @ WinB
    wmma_gemm_bf16<<<dim3(NPAD1 / 128, SEQ / 128), 256, 0, stream>>>(
        hB, WinB, zx, SEQ, DINPROJ, DMODEL, DMODEL, NPAD1, DINPROJ);
    // 3) dt / a / cumsum
    dtprep_kernel<<<NHEADS, 256, 0, stream>>>(zx, dt_bias, A_log, dtg, ag, cumg);
    // 4) serial scan
    scan_kernel<<<NHEADS, 256, 0, stream>>>(zx, dtg, ag, cumg, omega, Dp, yws);
    // 5) gate + norm -> bf16
    gate_norm_kernel<<<SEQ, 256, 0, stream>>>(zx, yws, nw, gB);
    // 6) GEMM2: y2[1024x768] = gB @ WoutB
    wmma_gemm_bf16<<<dim3(DMODEL / 128, SEQ / 128), 256, 0, stream>>>(
        gB, WoutB, y2, SEQ, DMODEL, DINNER, DINNER, DMODEL, DMODEL);
    // 7) post-norm + residual
    postnorm_kernel<<<SEQ, 256, 0, stream>>>(y2, x, post_scale, out);
}
